// AttnDecoder_37615323578981
// MI455X (gfx1250) — compile-verified, matrix operand order checked by reading, not run on hardware
//
#include <hip/hip_runtime.h>
#include <cstdint>

#define B_ 64
#define S_ 128
#define H_ 512
#define E_ 512
#define V_ 32000
#define T_ 30
#define SOS_ 2

typedef __attribute__((ext_vector_type(16))) __bf16 bf16x16;
typedef __attribute__((ext_vector_type(8)))  float  f32x8;

__device__ __forceinline__ unsigned short f2bf(float f) {
  union { float f; unsigned u; } x; x.f = f;
  unsigned r = x.u + 0x7FFFu + ((x.u >> 16) & 1u);
  return (unsigned short)(r >> 16);
}

// ---------------- generic utility kernels ----------------

__global__ void k_f32_to_bf16(const float* __restrict__ in,
                              unsigned short* __restrict__ out, long n) {
  long i = (long)blockIdx.x * blockDim.x + threadIdx.x;
  long st = (long)gridDim.x * blockDim.x;
  for (; i < n; i += st) out[i] = f2bf(in[i]);
}

__global__ void k_copy_f32(const float* __restrict__ in, float* __restrict__ out, int n) {
  int i = blockIdx.x * blockDim.x + threadIdx.x;
  if (i < n) out[i] = in[i];
}

// gather teacher-forced embeddings for all steps: out[t][b][e] (bf16)
__global__ void k_embed_gather(const float* __restrict__ emb, const int* __restrict__ tgt,
                               unsigned short* __restrict__ out) {
  long n = (long)T_ * B_ * E_;
  long st = (long)gridDim.x * blockDim.x;
  for (long i = (long)blockIdx.x * blockDim.x + threadIdx.x; i < n; i += st) {
    int e = (int)(i % E_);
    long tb = i / E_;
    int b = (int)(tb % B_);
    int t = (int)(tb / B_);
    int tok = (t == 0) ? SOS_ : tgt[b * T_ + (t - 1)];
    out[i] = f2bf(emb[(long)tok * E_ + e]);
  }
}

// ---------------- generic bf16 WMMA GEMM (one wave per 16x16 tile) ----------------
// C[M,N] = A[M,K] * B[N,K]^T. Used only for the one-time enc_ua precompute.
__global__ void k_gemm_bf16_wmma(const unsigned short* __restrict__ A,
                                 const unsigned short* __restrict__ Bm,
                                 const float* __restrict__ bias,
                                 float* __restrict__ C, int M, int N, int K) {
  int wave = threadIdx.x >> 5;
  int lane = threadIdx.x & 31;
  int tile = blockIdx.x * 4 + wave;
  int mtiles = M >> 4;
  int total = mtiles * (N >> 4);
  if (tile >= total) return;                 // wave-uniform -> EXEC stays all-ones
  int mt = tile % mtiles;
  int nt = tile / mtiles;

  // 16-bit fragment layout (16x32): lanes 0-15 hold row 0..15, K={0..7,16..23};
  // lanes 16-31 hold row 0..15, K={8..15,24..31}. B^T (N x K) mirrors A.
  int row   = lane & 15;
  int khalf = (lane >> 4) << 3;              // 0 or 8
  const unsigned short* ap = A  + (long)(mt * 16 + row) * K + khalf;
  const unsigned short* bp = Bm + (long)(nt * 16 + row) * K + khalf;

  f32x8 acc = {};
  for (int k = 0; k < K; k += 32) {
    union { bf16x16 v; uint4 q[2]; } a, b;
    a.q[0] = *(const uint4*)(ap + k);
    a.q[1] = *(const uint4*)(ap + k + 16);
    b.q[0] = *(const uint4*)(bp + k);
    b.q[1] = *(const uint4*)(bp + k + 16);
    acc = __builtin_amdgcn_wmma_f32_16x16x32_bf16(
        false, a.v, false, b.v, (short)0, acc, false, false);
  }

  int cn  = nt * 16 + (lane & 15);
  int cm0 = mt * 16 + ((lane >> 4) << 3);
  float bv = bias ? bias[cn] : 0.0f;
  for (int r = 0; r < 8; ++r)
    C[(long)(cm0 + r) * N + cn] = acc[r] + bv;
}

// ---------------- M=64 bf16 WMMA GEMM: one wave per N-tile, 4 M-tile accumulators --
// C[64,N] = A[64,K] * B[N,K]^T (+bias). B fragment loaded once per k-step and
// reused across 4 WMMAs -> weight stream read exactly once (the HBM roofline term).
__global__ void k_gemm_m64_wmma(const unsigned short* __restrict__ A,
                                const unsigned short* __restrict__ Bm,
                                const float* __restrict__ bias,
                                float* __restrict__ C, int N, int K) {
  int wave = threadIdx.x >> 5;
  int lane = threadIdx.x & 31;
  int nt = blockIdx.x * 4 + wave;
  if (nt >= (N >> 4)) return;                // wave-uniform

  int row   = lane & 15;
  int khalf = (lane >> 4) << 3;              // 0 or 8
  const unsigned short* ap = A  + (long)row * K + khalf;          // M-tile 0 base
  const unsigned short* bp = Bm + (long)(nt * 16 + row) * K + khalf;
  const long mstride = (long)16 * K;         // shorts between M-tiles

  f32x8 acc0 = {}, acc1 = {}, acc2 = {}, acc3 = {};
  for (int k = 0; k < K; k += 32) {
    union { bf16x16 v; uint4 q[2]; } b, a0, a1, a2, a3;
    b.q[0]  = *(const uint4*)(bp + k);
    b.q[1]  = *(const uint4*)(bp + k + 16);
    __builtin_prefetch(bp + k + 64, 0, 3);   // global_prefetch_b8 on the weight stream
    a0.q[0] = *(const uint4*)(ap + k);
    a0.q[1] = *(const uint4*)(ap + k + 16);
    a1.q[0] = *(const uint4*)(ap + mstride + k);
    a1.q[1] = *(const uint4*)(ap + mstride + k + 16);
    a2.q[0] = *(const uint4*)(ap + 2 * mstride + k);
    a2.q[1] = *(const uint4*)(ap + 2 * mstride + k + 16);
    a3.q[0] = *(const uint4*)(ap + 3 * mstride + k);
    a3.q[1] = *(const uint4*)(ap + 3 * mstride + k + 16);
    acc0 = __builtin_amdgcn_wmma_f32_16x16x32_bf16(false, a0.v, false, b.v, (short)0, acc0, false, false);
    acc1 = __builtin_amdgcn_wmma_f32_16x16x32_bf16(false, a1.v, false, b.v, (short)0, acc1, false, false);
    acc2 = __builtin_amdgcn_wmma_f32_16x16x32_bf16(false, a2.v, false, b.v, (short)0, acc2, false, false);
    acc3 = __builtin_amdgcn_wmma_f32_16x16x32_bf16(false, a3.v, false, b.v, (short)0, acc3, false, false);
  }

  int cn  = nt * 16 + (lane & 15);
  int cm0 = (lane >> 4) << 3;
  float bv = bias ? bias[cn] : 0.0f;
  for (int r = 0; r < 8; ++r) {
    C[(long)(cm0 + r) * N + cn]      = acc0[r] + bv;
    C[(long)(cm0 + 16 + r) * N + cn] = acc1[r] + bv;
    C[(long)(cm0 + 32 + r) * N + cn] = acc2[r] + bv;
    C[(long)(cm0 + 48 + r) * N + cn] = acc3[r] + bv;
  }
}

// ---------------- Bahdanau attention (one workgroup per batch row) ----------------
__global__ void k_attention(const float* __restrict__ hidden,   // [B,H]
                            const float* __restrict__ wa,       // [H,H]
                            const float* __restrict__ va,       // [H]
                            const float* __restrict__ enc_ua,   // [B,S,H]
                            const float* __restrict__ enc_out,  // [B,S,2H]
                            const unsigned short* __restrict__ emb_t, // [B,E] bf16
                            float* __restrict__ context,        // [B,2H]
                            unsigned short* __restrict__ x_bf,  // [B,E+2H] bf16
                            unsigned short* __restrict__ h_bf)  // [B,H]  bf16
{
  __shared__ float sh_hid[H_];
  __shared__ float sh_hwa[H_];
  __shared__ float sh_sc[S_];
  __shared__ float sh_red[2];
  int b = blockIdx.x, tid = threadIdx.x;
  int wave = tid >> 5, lane = tid & 31;

  for (int k = tid; k < H_; k += 256) {
    float hv = hidden[b * H_ + k];
    sh_hid[k] = hv;
    h_bf[b * H_ + k] = f2bf(hv);
  }
  __syncthreads();

  // hWa = hidden @ wa^T
  for (int h = tid; h < H_; h += 256) {
    const float* wr = wa + (long)h * H_;
    float s = 0.f;
    for (int k = 0; k < H_; ++k) s += sh_hid[k] * wr[k];
    sh_hwa[h] = s;
  }
  __syncthreads();

  // scores[s] = sum_h va[h] * tanh(hWa[h] + enc_ua[b,s,h])  (one wave per s)
  for (int s = wave; s < S_; s += 8) {
    const float* eu = enc_ua + ((long)b * S_ + s) * H_;
    float p = 0.f;
    for (int h = lane; h < H_; h += 32) p += va[h] * tanhf(sh_hwa[h] + eu[h]);
    for (int m = 16; m; m >>= 1) p += __shfl_xor(p, m, 32);
    if (lane == 0) sh_sc[s] = p;
  }
  __syncthreads();

  // softmax over S (wave 0 reduces)
  if (wave == 0) {
    float m = -1e30f;
    for (int s = lane; s < S_; s += 32) m = fmaxf(m, sh_sc[s]);
    for (int mm = 16; mm; mm >>= 1) m = fmaxf(m, __shfl_xor(m, mm, 32));
    float sum = 0.f;
    for (int s = lane; s < S_; s += 32) sum += expf(sh_sc[s] - m);
    for (int mm = 16; mm; mm >>= 1) sum += __shfl_xor(sum, mm, 32);
    if (lane == 0) { sh_red[0] = m; sh_red[1] = 1.0f / sum; }
  }
  __syncthreads();
  float mx = sh_red[0], inv = sh_red[1];
  for (int s = tid; s < S_; s += 256) sh_sc[s] = expf(sh_sc[s] - mx) * inv;
  __syncthreads();

  // context[d] = sum_s attw[s] * enc_out[b,s,d]
  for (int d = tid; d < 2 * H_; d += 256) {
    const float* eo = enc_out + (long)b * S_ * (2 * H_) + d;
    float c = 0.f;
    for (int s = 0; s < S_; ++s) c += sh_sc[s] * eo[(long)s * (2 * H_)];
    context[b * 2 * H_ + d] = c;
    x_bf[(long)b * (E_ + 2 * H_) + E_ + d] = f2bf(c);
  }
  for (int e = tid; e < E_; e += 256)
    x_bf[(long)b * (E_ + 2 * H_) + e] = emb_t[b * E_ + e];
}

// ---------------- GRU gate fusion + build fc input y = [emb, h_new, ctx] ----------------
__global__ void k_gru_gate(const float* __restrict__ gi, const float* __restrict__ gh,
                           const float* __restrict__ b_ih, const float* __restrict__ b_hh,
                           const float* __restrict__ context,
                           const unsigned short* __restrict__ emb_t,
                           float* __restrict__ hidden,
                           unsigned short* __restrict__ y_bf) // [B, E+3H]
{
  int i = blockIdx.x * blockDim.x + threadIdx.x;
  if (i >= B_ * H_) return;
  int b = i / H_, h = i % H_;
  const long g = (long)b * 3 * H_;
  float ir = gi[g + h]          + b_ih[h];
  float iz = gi[g + H_ + h]     + b_ih[H_ + h];
  float in_= gi[g + 2 * H_ + h] + b_ih[2 * H_ + h];
  float hr = gh[g + h]          + b_hh[h];
  float hz = gh[g + H_ + h]     + b_hh[H_ + h];
  float hn = gh[g + 2 * H_ + h] + b_hh[2 * H_ + h];
  float r = 1.f / (1.f + expf(-(ir + hr)));
  float z = 1.f / (1.f + expf(-(iz + hz)));
  float n = tanhf(in_ + r * hn);
  float hnew = (1.f - z) * n + z * hidden[i];
  hidden[i] = hnew;
  long yb = (long)b * (E_ + 3 * H_);
  y_bf[yb + h]                = emb_t[b * E_ + h];                // E_ == H_
  y_bf[yb + E_ + h]           = f2bf(hnew);
  y_bf[yb + E_ + H_ + h]      = f2bf(context[b * 2 * H_ + h]);
  y_bf[yb + E_ + 2 * H_ + h]  = f2bf(context[b * 2 * H_ + H_ + h]);
}

// ---------------- log-softmax over V, writing d_out[b, t, :] ----------------
__global__ void k_logsoftmax(const float* __restrict__ logits, float* __restrict__ out_t) {
  __shared__ float wred[8];
  __shared__ float red[2];
  int b = blockIdx.x, tid = threadIdx.x, lane = tid & 31, wave = tid >> 5;
  const float* row = logits + (long)b * V_;
  float m = -1e30f;
  for (int v = tid; v < V_; v += 256) m = fmaxf(m, row[v]);
  for (int mm = 16; mm; mm >>= 1) m = fmaxf(m, __shfl_xor(m, mm, 32));
  if (lane == 0) wred[wave] = m;
  __syncthreads();
  if (tid == 0) { float t = wred[0]; for (int i = 1; i < 8; ++i) t = fmaxf(t, wred[i]); red[0] = t; }
  __syncthreads();
  m = red[0];
  float s = 0.f;
  for (int v = tid; v < V_; v += 256) s += expf(row[v] - m);
  for (int mm = 16; mm; mm >>= 1) s += __shfl_xor(s, mm, 32);
  if (lane == 0) wred[wave] = s;
  __syncthreads();
  if (tid == 0) { float t = 0.f; for (int i = 0; i < 8; ++i) t += wred[i]; red[1] = logf(t); }
  __syncthreads();
  float lse = m + red[1];
  float* orow = out_t + (long)b * (T_ * V_);
  for (int v = tid; v < V_; v += 256) orow[v] = row[v] - lse;
}

// ---------------- host orchestration ----------------
extern "C" void kernel_launch(void* const* d_in, const int* in_sizes, int n_in,
                              void* d_out, int out_size, void* d_ws, size_t ws_size,
                              hipStream_t stream) {
  const float* enc_out = (const float*)d_in[0];
  const float* enc_hid = (const float*)d_in[1];
  const int*   target  = (const int*)d_in[2];
  const float* emb     = (const float*)d_in[3];
  const float* wa      = (const float*)d_in[4];
  const float* ua      = (const float*)d_in[5];
  const float* va      = (const float*)d_in[6];
  const float* w_ih    = (const float*)d_in[7];
  const float* w_hh    = (const float*)d_in[8];
  const float* b_ih    = (const float*)d_in[9];
  const float* b_hh    = (const float*)d_in[10];
  const float* fc_w    = (const float*)d_in[11];
  const float* fc_b    = (const float*)d_in[12];
  float* out = (float*)d_out;

  char* ws = (char*)d_ws;
  size_t off = 0;
  auto alloc = [&](size_t bytes) -> void* {
    void* p = (void*)(ws + off);
    off += (bytes + 255) & ~(size_t)255;
    return p;
  };
  unsigned short* enc_out_bf = (unsigned short*)alloc((size_t)B_ * S_ * 2 * H_ * 2);
  unsigned short* ua_bf      = (unsigned short*)alloc((size_t)H_ * 2 * H_ * 2);
  unsigned short* w_ih_bf    = (unsigned short*)alloc((size_t)3 * H_ * (E_ + 2 * H_) * 2);
  unsigned short* w_hh_bf    = (unsigned short*)alloc((size_t)3 * H_ * H_ * 2);
  unsigned short* fc_w_bf    = (unsigned short*)alloc((size_t)V_ * (E_ + 3 * H_) * 2);
  unsigned short* emb_all    = (unsigned short*)alloc((size_t)T_ * B_ * E_ * 2);
  float*          enc_ua     = (float*)alloc((size_t)B_ * S_ * H_ * 4);
  float*          hidden     = (float*)alloc((size_t)B_ * H_ * 4);
  unsigned short* hid_bf     = (unsigned short*)alloc((size_t)B_ * H_ * 2);
  float*          context    = (float*)alloc((size_t)B_ * 2 * H_ * 4);
  unsigned short* x_bf       = (unsigned short*)alloc((size_t)B_ * (E_ + 2 * H_) * 2);
  unsigned short* y_bf       = (unsigned short*)alloc((size_t)B_ * (E_ + 3 * H_) * 2);
  float*          gi         = (float*)alloc((size_t)B_ * 3 * H_ * 4);
  float*          gh         = (float*)alloc((size_t)B_ * 3 * H_ * 4);
  float*          logits     = (float*)alloc((size_t)B_ * V_ * 4);

  // one-time: bf16 weight conversions + embedding gather + enc_ua GEMM
  k_f32_to_bf16<<<2048, 256, 0, stream>>>(enc_out, enc_out_bf, (long)B_ * S_ * 2 * H_);
  k_f32_to_bf16<<<512, 256, 0, stream>>>(ua, ua_bf, (long)H_ * 2 * H_);
  k_f32_to_bf16<<<1024, 256, 0, stream>>>(w_ih, w_ih_bf, (long)3 * H_ * (E_ + 2 * H_));
  k_f32_to_bf16<<<512, 256, 0, stream>>>(w_hh, w_hh_bf, (long)3 * H_ * H_);
  k_f32_to_bf16<<<4096, 256, 0, stream>>>(fc_w, fc_w_bf, (long)V_ * (E_ + 3 * H_));
  k_embed_gather<<<1024, 256, 0, stream>>>(emb, target, emb_all);
  k_copy_f32<<<(B_ * H_ + 255) / 256, 256, 0, stream>>>(enc_hid, hidden, B_ * H_);

  {   // enc_ua[b,s,h] = sum_k enc_out[b,s,k] * ua[h,k]  : M=8192, N=512, K=1024
    int tiles = (B_ * S_ / 16) * (H_ / 16);
    k_gemm_bf16_wmma<<<(tiles + 3) / 4, 128, 0, stream>>>(
        enc_out_bf, ua_bf, nullptr, enc_ua, B_ * S_, H_, 2 * H_);
  }

  for (int t = 0; t < T_; ++t) {
    const unsigned short* emb_t = emb_all + (size_t)t * B_ * E_;
    k_attention<<<B_, 256, 0, stream>>>(hidden, wa, va, enc_ua, enc_out,
                                        emb_t, context, x_bf, hid_bf);
    {   // gi = x @ w_ih^T : M=64, N=1536, K=1536 -> 96 N-tiles
      int ntiles = 3 * H_ / 16;
      k_gemm_m64_wmma<<<(ntiles + 3) / 4, 128, 0, stream>>>(
          x_bf, w_ih_bf, nullptr, gi, 3 * H_, E_ + 2 * H_);
    }
    {   // gh = hidden @ w_hh^T : M=64, N=1536, K=512 -> 96 N-tiles
      int ntiles = 3 * H_ / 16;
      k_gemm_m64_wmma<<<(ntiles + 3) / 4, 128, 0, stream>>>(
          hid_bf, w_hh_bf, nullptr, gh, 3 * H_, H_);
    }
    k_gru_gate<<<(B_ * H_ + 255) / 256, 256, 0, stream>>>(
        gi, gh, b_ih, b_hh, context, emb_t, hidden, y_bf);
    {   // logits = y @ fc_w^T + fc_b : M=64, N=32000, K=2048 -> 2000 N-tiles
      int ntiles = V_ / 16;
      k_gemm_m64_wmma<<<(ntiles + 3) / 4, 128, 0, stream>>>(
          y_bf, fc_w_bf, fc_b, logits, V_, E_ + 3 * H_);
    }
    k_logsoftmax<<<B_, 256, 0, stream>>>(logits, out + (size_t)t * V_);
  }
}
